// GPS_8701603742428
// MI455X (gfx1250) — matrix operand order repaired and checked by reading, not compiled, not run on hardware
//
#include <hip/hip_runtime.h>
#include <hip/hip_bf16.h>

#define NN 4096   // nodes
#define EE 65536  // edges
#define CC 256    // channels
#define LL 3      // layers
#define HH 8      // heads
#define DH 32     // head dim

typedef __attribute__((ext_vector_type(4)))  float  v4f;
typedef __attribute__((ext_vector_type(8)))  float  v8f;
typedef __attribute__((ext_vector_type(16))) __bf16 v16bf;

union Frag16 { v4f f4[2]; v16bf v; };

// ---------------- elementwise / conversion kernels ----------------

__global__ __launch_bounds__(256) void lin1_kernel(
    const float* __restrict__ x, const float* __restrict__ w,
    const float* __restrict__ b, float* __restrict__ hF, __bf16* __restrict__ hB) {
  int i = blockIdx.x * 256 + threadIdx.x;     // NN*CC threads
  int n = i >> 8, c = i & (CC - 1);
  float v = x[n] * w[c] + b[c];
  hF[i] = v;
  hB[i] = (__bf16)v;
}

__global__ __launch_bounds__(256) void copy_f32_kernel(
    const float* __restrict__ in, float* __restrict__ out, int n) {
  int i = blockIdx.x * 256 + threadIdx.x;
  if (i < n) out[i] = in[i];
}

__global__ __launch_bounds__(256) void conv_bf16_kernel(
    const float* __restrict__ in, __bf16* __restrict__ out, int n, float scale) {
  int i = blockIdx.x * 256 + threadIdx.x;
  if (i < n) out[i] = (__bf16)(in[i] * scale);
}

__global__ __launch_bounds__(256) void add_conv_kernel(
    const float* __restrict__ a, const float* __restrict__ b,
    float* __restrict__ outF, __bf16* __restrict__ outB, int n) {
  int i = blockIdx.x * 256 + threadIdx.x;
  if (i < n) {
    float v = a[i] + b[i];
    outF[i] = v;
    outB[i] = (__bf16)v;
  }
}

// [NN][CC] f32 -> [CC][NN] bf16
__global__ __launch_bounds__(256) void transpose_conv_kernel(
    const float* __restrict__ in, __bf16* __restrict__ out) {
  int i = blockIdx.x * 256 + threadIdx.x;     // NN*CC threads
  int n = i & (NN - 1), c = i >> 12;
  out[(size_t)c * NN + n] = (__bf16)in[(size_t)n * CC + c];
}

// ---------------- GIN neighbor aggregation (scatter-add) ----------------

__global__ __launch_bounds__(256) void gin_scatter_kernel(
    const float* __restrict__ h, const int* __restrict__ ei, float* __restrict__ agg) {
  long i = (long)blockIdx.x * 256 + threadIdx.x;   // EE * (CC/4) threads
  int e = (int)(i >> 6);
  int c4 = (int)(i & 63) << 2;
  int s = ei[e], d = ei[EE + e];
  const float* sp = h + (size_t)s * CC + c4;
  float* dp = agg + (size_t)d * CC + c4;
  atomicAdd(dp + 0, sp[0]);
  atomicAdd(dp + 1, sp[1]);
  atomicAdd(dp + 2, sp[2]);
  atomicAdd(dp + 3, sp[3]);
}

// ---------------- WMMA GEMM: out = act(A @ W^T + bias) ----------------
// A: [M][K] bf16 row-major, W: [N][K] bf16 row-major (so B = W^T).
// Each wave computes a 16x64 output tile (4 accumulators).
// K-loop manually unrolled 2x with two *named* fragment buffers so all local
// arrays are constant-indexed (SROA keeps them in VGPRs, no scratch spills)
// while loads for one buffer overlap the other buffer's WMMA chain.
// Requires K % 64 == 0 (true here: 256, 512).

__global__ __launch_bounds__(256) void gemm_bf16_kernel(
    const __bf16* __restrict__ A, const __bf16* __restrict__ W,
    const float* __restrict__ bias, float* __restrict__ outF,
    __bf16* __restrict__ outB, int M, int N, int K, int relu, float bscale) {
  const int wave = blockIdx.x * 8 + (threadIdx.x >> 5);
  const int lane = threadIdx.x & 31;
  const int nt = N >> 6;
  const int tm = wave / nt, tn = wave % nt;
  const int half = lane >> 4, l16 = lane & 15;

  const __bf16* abase = A + (size_t)(tm * 16 + l16) * K;
  const __bf16* wbase0 = W + (size_t)(tn * 64 + l16) * K + half * 16;

  v8f acc[4] = {};
  Frag16 af0, af1;
  Frag16 b0[4], b1[4];

  auto loadA = [&](int kb, Frag16& a) {
    a.f4[0] = *(const v4f*)(abase + kb + half * 16);       // e0..7:  k = kb+half*16+o
    a.f4[1] = *(const v4f*)(abase + kb + half * 16 + 8);  // e8..15: k = kb+half*16+8+o
  };
  auto loadB = [&](int kb, Frag16* b) {
#pragma unroll
    for (int t = 0; t < 4; t++) {
      const __bf16* wb = wbase0 + (size_t)(t * 16) * K + kb;
      b[t].f4[0] = *(const v4f*)wb;                       // k = kb+half*16+e
      b[t].f4[1] = *(const v4f*)(wb + 8);
    }
  };

  loadA(0, af0);
  loadB(0, b0);

  for (int kb = 0; kb < K; kb += 64) {
    // stage buffer 1 (kb+32) while buffer 0 computes
    loadA(kb + 32, af1);
    loadB(kb + 32, b1);
    __builtin_prefetch((const void*)(abase + kb + 64), 0, 3);
#pragma unroll
    for (int t = 0; t < 4; t++)
      acc[t] = __builtin_amdgcn_wmma_f32_16x16x32_bf16(
          false, af0.v, false, b0[t].v, (short)0, acc[t], false, false);
    // stage buffer 0 (kb+64) while buffer 1 computes
    if (kb + 64 < K) {
      loadA(kb + 64, af0);
      loadB(kb + 64, b0);
    }
#pragma unroll
    for (int t = 0; t < 4; t++)
      acc[t] = __builtin_amdgcn_wmma_f32_16x16x32_bf16(
          false, af1.v, false, b1[t].v, (short)0, acc[t], false, false);
  }
#pragma unroll
  for (int t = 0; t < 4; t++) {
    int gcol = tn * 64 + t * 16 + l16;
    float bv = bias[gcol];
#pragma unroll
    for (int r = 0; r < 8; r++) {
      int gm = tm * 16 + half * 8 + r;                   // C/D row layout
      float v = acc[t][r] + bv;
      if (relu) v = fmaxf(v, 0.f);
      size_t o = (size_t)gm * N + gcol;
      if (outF) outF[o] = v;
      if (outB) outB[o] = (__bf16)(v * bscale);
    }
  }
}

// ---------------- Flash attention with async K/V staging ----------------
// All 8 waves of a block share one head (hd = wave>>8, 8 | 256), so K/V tiles
// are staged cooperatively into LDS with global_load_async_to_lds_b64,
// double-buffered (issue next tile, s_wait_asynccnt on current, barrier).
// Q: [NN][CC] bf16 (pre-scaled by 1/sqrt(dh)); K: [NN][CC] bf16;
// Vt: [CC][NN] bf16; O: [NN][CC] f32.

__global__ __launch_bounds__(256) void flash_attn_kernel(
    const __bf16* __restrict__ Q, const __bf16* __restrict__ Km,
    const __bf16* __restrict__ Vt, float* __restrict__ O) {
  __shared__ __bf16 kbuf[2][32 * 32];   // [key][d]  2KB per buffer
  __shared__ __bf16 vbuf[2][32 * 32];   // [d][key]  2KB per buffer
  __shared__ __bf16 pstage[8][16 * 32]; // per-wave P tile

  const int tid = threadIdx.x;
  const int wv = tid >> 5;
  const int lane = tid & 31;
  const int wave = blockIdx.x * 8 + wv;
  const int hd = wave >> 8;        // uniform within the block
  const int qt = wave & 255;
  const int half = lane >> 4, l16 = lane & 15;

  // cooperative staging: 256 threads x 8B cover one 2KB tile each for K and V
  const int srow = tid >> 3;            // 0..31
  const int scol = (tid & 7) << 3;      // byte column 0..56
  const unsigned klds0 = (unsigned)(size_t)(&kbuf[0][0]);
  const unsigned klds1 = (unsigned)(size_t)(&kbuf[1][0]);
  const unsigned vlds0 = (unsigned)(size_t)(&vbuf[0][0]);
  const unsigned vlds1 = (unsigned)(size_t)(&vbuf[1][0]);
  const unsigned kg_base = (unsigned)((srow * CC + hd * DH) * 2 + scol);
  const unsigned vg_base = (unsigned)(((hd * DH + srow) * NN) * 2 + scol);

  auto stage = [&](int kb, int buf) {
    unsigned gk = kg_base + (unsigned)kb * (CC * 2);
    unsigned gv = vg_base + (unsigned)kb * 2;
    unsigned lk = (buf ? klds1 : klds0) + (unsigned)(srow * 64 + scol);
    unsigned lv = (buf ? vlds1 : vlds0) + (unsigned)(srow * 64 + scol);
    asm volatile("global_load_async_to_lds_b64 %0, %1, %2"
                 :: "v"(lk), "v"(gk), "s"(Km) : "memory");
    asm volatile("global_load_async_to_lds_b64 %0, %1, %2"
                 :: "v"(lv), "v"(gv), "s"(Vt) : "memory");
  };

  Frag16 qf;
  const __bf16* qbase = Q + (size_t)(qt * 16 + l16) * CC + hd * DH;
  qf.f4[0] = *(const v4f*)(qbase + half * 16);
  qf.f4[1] = *(const v4f*)(qbase + half * 16 + 8);

  v8f o0 = {}, o1 = {};
  float Mrow[8], Srow[8];
#pragma unroll
  for (int r = 0; r < 8; r++) { Mrow[r] = -1e30f; Srow[r] = 0.f; }

  __bf16* pp = pstage[wv];

  stage(0, 0);

  for (int kb = 0; kb < NN; kb += 32) {
    const int cur = (kb >> 5) & 1;
    if (kb + 32 < NN) {
      stage(kb + 32, cur ^ 1);
      asm volatile("s_wait_asynccnt 0x2" ::: "memory");   // current tile done
    } else {
      asm volatile("s_wait_asynccnt 0x0" ::: "memory");
    }
    __syncthreads();                                      // tile visible to all

    const __bf16* kl = cur ? kbuf[1] : kbuf[0];
    const __bf16* vl = cur ? vbuf[1] : vbuf[0];

    // --- scores: two 16-key tiles, one WMMA each (K-depth = dh = 32) ---
    v8f s0 = {}, s1 = {};
    Frag16 bf, bg;
    bf.f4[0] = *(const v4f*)(kl + l16 * 32 + half * 16);
    bf.f4[1] = *(const v4f*)(kl + l16 * 32 + half * 16 + 8);
    bg.f4[0] = *(const v4f*)(kl + (16 + l16) * 32 + half * 16);
    bg.f4[1] = *(const v4f*)(kl + (16 + l16) * 32 + half * 16 + 8);
    s0 = __builtin_amdgcn_wmma_f32_16x16x32_bf16(false, qf.v, false, bf.v, (short)0, s0, false, false);
    s1 = __builtin_amdgcn_wmma_f32_16x16x32_bf16(false, qf.v, false, bg.v, (short)0, s1, false, false);

    // V fragments fetched now (before barrier B), consumed after it
    Frag16 vf, vg;
    vf.f4[0] = *(const v4f*)(vl + l16 * 32 + half * 16);
    vf.f4[1] = *(const v4f*)(vl + l16 * 32 + half * 16 + 8);
    vg.f4[0] = *(const v4f*)(vl + (16 + l16) * 32 + half * 16);
    vg.f4[1] = *(const v4f*)(vl + (16 + l16) * 32 + half * 16 + 8);

    // --- online softmax; row r lives in lanes of this half-wave ---
#pragma unroll
    for (int r = 0; r < 8; r++) {
      float mx = fmaxf(s0[r], s1[r]);
      mx = fmaxf(mx, __shfl_xor(mx, 1, 32));
      mx = fmaxf(mx, __shfl_xor(mx, 2, 32));
      mx = fmaxf(mx, __shfl_xor(mx, 4, 32));
      mx = fmaxf(mx, __shfl_xor(mx, 8, 32));
      mx = fmaxf(mx, Mrow[r]);
      float p0 = __expf(s0[r] - mx), p1 = __expf(s1[r] - mx);
      float rs = p0 + p1;
      rs += __shfl_xor(rs, 1, 32);
      rs += __shfl_xor(rs, 2, 32);
      rs += __shfl_xor(rs, 4, 32);
      rs += __shfl_xor(rs, 8, 32);
      float cr = __expf(Mrow[r] - mx);
      Srow[r] = Srow[r] * cr + rs;
      Mrow[r] = mx;
      o0[r] *= cr; o1[r] *= cr;
      s0[r] = p0; s1[r] = p1;
    }
    // --- stage P (16 rows x 32 keys) through LDS to re-fragment as A ---
#pragma unroll
    for (int r = 0; r < 8; r++) {
      int row = half * 8 + r;
      pp[row * 32 + l16] = (__bf16)s0[r];
      pp[row * 32 + 16 + l16] = (__bf16)s1[r];
    }
    // All reads of the current K/V buffer (incl. V fragments) must complete
    // before any wave can issue next-next async writes into this buffer.
    asm volatile("s_wait_dscnt 0x0" ::: "memory");
    __syncthreads();
    Frag16 pf;
    pf.f4[0] = *(const v4f*)(pp + l16 * 32 + half * 16);
    pf.f4[1] = *(const v4f*)(pp + l16 * 32 + half * 16 + 8);
    // --- O += P @ V_block ---
    o0 = __builtin_amdgcn_wmma_f32_16x16x32_bf16(false, pf.v, false, vf.v, (short)0, o0, false, false);
    o1 = __builtin_amdgcn_wmma_f32_16x16x32_bf16(false, pf.v, false, vg.v, (short)0, o1, false, false);
  }
#pragma unroll
  for (int r = 0; r < 8; r++) {
    float inv = 1.0f / Srow[r];
    int gm = qt * 16 + half * 8 + r;
    O[(size_t)gm * CC + hd * DH + l16] = o0[r] * inv;
    O[(size_t)gm * CC + hd * DH + 16 + l16] = o1[r] * inv;
  }
}

// ---------------- BatchNorm (training mode, biased var) with fused residual ----------------

__global__ __launch_bounds__(256) void bn_kernel(
    const float* __restrict__ A, const float* __restrict__ Badd,
    const float* __restrict__ g, const float* __restrict__ b,
    float* __restrict__ outF, __bf16* __restrict__ outB) {
  int c = blockIdx.x;
  int tid = threadIdx.x;
  __shared__ float s1[256], s2[256];
  float sum = 0.f, sq = 0.f;
  for (int n = tid; n < NN; n += 256) {
    float v = A[(size_t)n * CC + c];
    if (Badd) v += Badd[(size_t)n * CC + c];
    sum += v; sq += v * v;
  }
  s1[tid] = sum; s2[tid] = sq;
  __syncthreads();
  for (int off = 128; off > 0; off >>= 1) {
    if (tid < off) { s1[tid] += s1[tid + off]; s2[tid] += s2[tid + off]; }
    __syncthreads();
  }
  float mean = s1[0] / (float)NN;
  float var = s2[0] / (float)NN - mean * mean;
  float rstd = rsqrtf(var + 1e-5f);
  float gg = g[c], bb = b[c];
  for (int n = tid; n < NN; n += 256) {
    float v = A[(size_t)n * CC + c];
    if (Badd) v += Badd[(size_t)n * CC + c];
    float o = (v - mean) * rstd * gg + bb;
    outF[(size_t)n * CC + c] = o;
    if (outB) outB[(size_t)n * CC + c] = (__bf16)o;
  }
}

// ---------------- host orchestration ----------------

extern "C" void kernel_launch(void* const* d_in, const int* in_sizes, int n_in,
                              void* d_out, int out_size, void* d_ws, size_t ws_size,
                              hipStream_t stream) {
  (void)in_sizes; (void)n_in; (void)out_size; (void)ws_size;
  const float* x       = (const float*)d_in[0];
  const int*   ei      = (const int*)d_in[1];
  const float* lin1_w  = (const float*)d_in[2];
  const float* lin1_b  = (const float*)d_in[3];
  const float* gin_w1  = (const float*)d_in[4];
  const float* gin_b1  = (const float*)d_in[5];
  const float* gin_w2  = (const float*)d_in[6];
  const float* gin_b2  = (const float*)d_in[7];
  const float* attn_wq = (const float*)d_in[8];
  const float* attn_wk = (const float*)d_in[9];
  const float* attn_wv = (const float*)d_in[10];
  const float* attn_wo = (const float*)d_in[11];
  const float* attn_bq = (const float*)d_in[12];
  const float* attn_bk = (const float*)d_in[13];
  const float* attn_bv = (const float*)d_in[14];
  const float* attn_bo = (const float*)d_in[15];
  const float* bn1_g   = (const float*)d_in[16];
  const float* bn1_b   = (const float*)d_in[17];
  const float* bn2_g   = (const float*)d_in[18];
  const float* bn2_b   = (const float*)d_in[19];
  const float* bn3_g   = (const float*)d_in[20];
  const float* bn3_b   = (const float*)d_in[21];
  const float* mlp_w1  = (const float*)d_in[22];
  const float* mlp_b1  = (const float*)d_in[23];
  const float* mlp_w2  = (const float*)d_in[24];
  const float* mlp_b2  = (const float*)d_in[25];

  constexpr size_t MB = (size_t)1 << 20;
  char* ws = (char*)d_ws;
  float*  hF   = (float*)(ws + 0 * MB);    // [NN][CC]
  __bf16* hB   = (__bf16*)(ws + 4 * MB);
  float*  F1   = (float*)(ws + 6 * MB);
  float*  F2   = (float*)(ws + 10 * MB);
  float*  F3   = (float*)(ws + 14 * MB);
  float*  F4   = (float*)(ws + 18 * MB);
  __bf16* B1   = (__bf16*)(ws + 22 * MB);
  __bf16* B2   = (__bf16*)(ws + 24 * MB);
  __bf16* B3   = (__bf16*)(ws + 26 * MB);  // Vt [CC][NN]
  __bf16* B2C  = (__bf16*)(ws + 28 * MB);  // [NN][2*CC]
  __bf16* wbuf = (__bf16*)(ws + 32 * MB);  // bf16 weights, <=2C*C

  const float scale = 0.17677669529663687f;  // 1/sqrt(DH)
  const int NC_BLKS = NN * CC / 256;

  auto gemm = [&](const __bf16* A, const float* Wf, const float* bias,
                  float* outF, __bf16* outB, int M, int N, int K,
                  int relu, float bscale) {
    int cnt = N * K;
    conv_bf16_kernel<<<(cnt + 255) / 256, 256, 0, stream>>>(Wf, wbuf, cnt, 1.f);
    int waves = (M / 16) * (N / 64);
    gemm_bf16_kernel<<<waves / 8, 256, 0, stream>>>(A, wbuf, bias, outF, outB,
                                                    M, N, K, relu, bscale);
  };

  // input projection: h = x @ lin1_w^T + b   (K=1, outer product)
  lin1_kernel<<<NC_BLKS, 256, 0, stream>>>(x, lin1_w, lin1_b, hF, hB);

  for (int l = 0; l < LL; l++) {
    const float* gw1 = gin_w1 + (size_t)l * CC * CC;
    const float* gb1 = gin_b1 + (size_t)l * CC;
    const float* gw2 = gin_w2 + (size_t)l * CC * CC;
    const float* gb2 = gin_b2 + (size_t)l * CC;
    const float* wq = attn_wq + (size_t)l * CC * CC;
    const float* wk = attn_wk + (size_t)l * CC * CC;
    const float* wv = attn_wv + (size_t)l * CC * CC;
    const float* wo = attn_wo + (size_t)l * CC * CC;
    const float* bq = attn_bq + (size_t)l * CC;
    const float* bk = attn_bk + (size_t)l * CC;
    const float* bv = attn_bv + (size_t)l * CC;
    const float* bo = attn_bo + (size_t)l * CC;
    const float* mw1 = mlp_w1 + (size_t)l * 2 * CC * CC;
    const float* mb1 = mlp_b1 + (size_t)l * 2 * CC;
    const float* mw2 = mlp_w2 + (size_t)l * 2 * CC * CC;
    const float* mb2 = mlp_b2 + (size_t)l * CC;

    // --- GIN branch: agg = h + scatter_sum(h[src] -> dst) ---
    copy_f32_kernel<<<NC_BLKS, 256, 0, stream>>>(hF, F1, NN * CC);
    gin_scatter_kernel<<<EE * (CC / 4) / 256, 256, 0, stream>>>(hF, ei, F1);
    conv_bf16_kernel<<<NC_BLKS, 256, 0, stream>>>(F1, B1, NN * CC, 1.f);
    gemm(B1, gw1, gb1, nullptr, B2, NN, CC, CC, 1, 1.f);     // relu(agg@W1^T+b1)
    gemm(B2, gw2, gb2, F3, nullptr, NN, CC, CC, 0, 1.f);     // z
    bn_kernel<<<CC, 256, 0, stream>>>(F3, hF, bn1_g + l * CC, bn1_b + l * CC,
                                      F4, nullptr);          // h1 = bn(z + h)

    // --- global attention ---
    gemm(hB, wq, bq, nullptr, B1, NN, CC, CC, 0, scale);     // qB (pre-scaled)
    gemm(hB, wk, bk, nullptr, B2, NN, CC, CC, 0, 1.f);       // kB
    gemm(hB, wv, bv, F1, nullptr, NN, CC, CC, 0, 1.f);       // v f32
    transpose_conv_kernel<<<NC_BLKS, 256, 0, stream>>>(F1, B3);  // Vt bf16
    flash_attn_kernel<<<HH * (NN / 16) / 8, 256, 0, stream>>>(B1, B2, B3, F2);
    conv_bf16_kernel<<<NC_BLKS, 256, 0, stream>>>(F2, B1, NN * CC, 1.f);
    gemm(B1, wo, bo, F3, nullptr, NN, CC, CC, 0, 1.f);       // o @ Wo^T + bo
    bn_kernel<<<CC, 256, 0, stream>>>(F3, hF, bn2_g + l * CC, bn2_b + l * CC,
                                      F2, nullptr);          // h2

    // --- combine + feedforward ---
    add_conv_kernel<<<NC_BLKS, 256, 0, stream>>>(F4, F2, F1, B1, NN * CC);  // s
    gemm(B1, mw1, mb1, nullptr, B2C, NN, 2 * CC, CC, 1, 1.f);
    gemm(B2C, mw2, mb2, F2, nullptr, NN, CC, 2 * CC, 0, 1.f);
    bool last = (l == LL - 1);
    bn_kernel<<<CC, 256, 0, stream>>>(F2, F1, bn3_g + l * CC, bn3_b + l * CC,
                                      last ? (float*)d_out : hF,
                                      last ? nullptr : hB);  // h = bn(s + m)
  }
}
